// LSTMSoftAttentionNoInputCore_61735859913415
// MI455X (gfx1250) — matrix operand order repaired and checked by reading, not compiled, not run on hardware
//
#include <hip/hip_runtime.h>
#include <hip/hip_bf16.h>
#include <math.h>

// ---------------- types ----------------
typedef __attribute__((ext_vector_type(16))) __bf16 v16bf;
typedef __attribute__((ext_vector_type(8)))  __bf16 v8bf;
typedef __attribute__((ext_vector_type(8)))  float  v8f;
typedef __attribute__((ext_vector_type(4)))  unsigned int u32x4;
typedef __attribute__((ext_vector_type(8)))  int    i32x8;
typedef __attribute__((ext_vector_type(4)))  int    i32x4;

// Problem constants
#define Bsz 128
#define Nsq 196
#define NPAD 208           // 13 * 16
#define Fdim 2048
#define Rdim 1024
#define Hdim 512

// TDM staging geometry for k_scores
#define KC     128         // fp32 elements per A-chunk along K
#define LDSROW 132         // 128 + 4 pad DWORDs (TDM pad: kills 64-bank conflicts)

// ---- TDM availability / arity discrimination (probe-verified difference) ----
#if defined(__has_builtin)
#if __has_builtin(__builtin_amdgcn_tensor_load_to_lds) && __has_builtin(__builtin_amdgcn_s_wait_tensorcnt)
#define HAVE_TDM 1
#endif
#endif
#ifndef HAVE_TDM
#define HAVE_TDM 0
#endif

__device__ __forceinline__ v8f zero_v8f() {
  v8f z;
#pragma unroll
  for (int i = 0; i < 8; ++i) z[i] = 0.0f;
  return z;
}

__device__ __forceinline__ v8f wmma_bf16(v16bf a, v16bf b, v8f c) {
  // v_wmma_f32_16x16x32_bf16 : D = A(16x32) * B(32x16) + C
  return __builtin_amdgcn_wmma_f32_16x16x32_bf16(
      /*neg_a=*/false, a, /*neg_b=*/false, b,
      /*c_mod=*/(short)0, c, /*reuse_a=*/false, /*reuse_b=*/false);
}

// pack 16 fp32 (two 8-element K-groups) into a bf16 A fragment
__device__ __forceinline__ v16bf pack16(float4 f0, float4 f1, float4 g0, float4 g1) {
  v16bf a;
  a[0] = (__bf16)f0.x;  a[1] = (__bf16)f0.y;  a[2]  = (__bf16)f0.z;  a[3]  = (__bf16)f0.w;
  a[4] = (__bf16)f1.x;  a[5] = (__bf16)f1.y;  a[6]  = (__bf16)f1.z;  a[7]  = (__bf16)f1.w;
  a[8] = (__bf16)g0.x;  a[9] = (__bf16)g0.y;  a[10] = (__bf16)g0.z;  a[11] = (__bf16)g0.w;
  a[12]= (__bf16)g1.x;  a[13]= (__bf16)g1.y;  a[14] = (__bf16)g1.z;  a[15] = (__bf16)g1.w;
  return a;
}

// A fragment (16x32 bf16) from a bf16 row-major matrix [rows][lda]
// ISA layout: lane<16 -> M=lane, K = {0..7, 16..23}; lane>=16 -> M=lane-16, K = {8..15, 24..31}
__device__ __forceinline__ v16bf load_a_bf(const __bf16* __restrict__ A, int lda,
                                           int m0, int k0, int lane) {
  int m  = m0 + (lane & 15);
  int kb = k0 + ((lane >> 4) << 3);          // 0 or 8
  const __bf16* p = A + (size_t)m * lda + kb;
  v8bf lo = *(const v8bf*)p;                 // K = kb .. kb+7
  v8bf hi = *(const v8bf*)(p + 16);          // K = kb+16 .. kb+23
  v16bf a;
#pragma unroll
  for (int e = 0; e < 8; ++e) { a[e] = lo[e]; a[8 + e] = hi[e]; }
  return a;
}

// A fragment from fp32 global memory (row clamped), converting to bf16
__device__ __forceinline__ v16bf load_a_f32(const float* __restrict__ A, int lda,
                                            int m0, int mmax, int k0, int lane) {
  int m = m0 + (lane & 15);
  if (m > mmax) m = mmax;
  int kb = k0 + ((lane >> 4) << 3);
  const float* p = A + (size_t)m * lda + kb;
  return pack16(*(const float4*)(p + 0), *(const float4*)(p + 4),
                *(const float4*)(p + 16), *(const float4*)(p + 20));
}

// A fragment from an fp32 LDS chunk [16][LDSROW]
__device__ __forceinline__ v16bf load_a_lds(const float* __restrict__ buf, int kk, int lane) {
  int m  = lane & 15;
  int kb = kk + ((lane >> 4) << 3);
  const float* p = buf + m * LDSROW + kb;
  return pack16(*(const float4*)(p + 0), *(const float4*)(p + 4),
                *(const float4*)(p + 16), *(const float4*)(p + 20));
}

// B fragment (32x16 bf16), B = W^T with W row-major [ncols][ldw]:
// lane<16 -> N=lane, K = {0..7, 16..23}; lane>=16 -> N=lane-16, K = {8..15, 24..31}
__device__ __forceinline__ v16bf load_b_bf(const __bf16* __restrict__ W, int ldw,
                                           int n0, int k0, int lane) {
  int n  = n0 + (lane & 15);
  int kb = k0 + ((lane >> 4) << 3);          // 0 or 8
  const __bf16* p = W + (size_t)n * ldw + kb;
  v8bf lo = *(const v8bf*)p;                 // K = kb .. kb+7
  v8bf hi = *(const v8bf*)(p + 16);          // K = kb+16 .. kb+23
  v16bf bv;
#pragma unroll
  for (int e = 0; e < 8; ++e) { bv[e] = lo[e]; bv[8 + e] = hi[e]; }
  return bv;
}

#if HAVE_TDM
// Issue one TDM 2D tile load: 16 rows x KC fp32 from global -> LDS with row padding.
// D# per CDNA5 ISA (2D: groups 2/3 zero).
__device__ __forceinline__ void tdm_issue(const float* gtile, unsigned lds_off, int rows_avail) {
  unsigned long long ga = (unsigned long long)(size_t)gtile;
  u32x4 g0 = { 1u,                                        // count=1 (valid user descriptor)
               lds_off,                                   // lds_addr (bytes)
               (unsigned)(ga & 0xFFFFFFFFu),              // global_addr[31:0]
               (unsigned)(((ga >> 32) & 0x01FFFFFFu) | (2u << 30)) };  // addr[56:32] | type=2
  i32x8 g1 = { (int)((2u << 16)                          // data_size = 4B
                     | (1u << 20)                        // pad_enable
                     | (6u << 22)                        // pad_interval: 128 DWORDs
                     | (3u << 25)),                      // pad_amount: 4 DWORDs -> 132/row
               (int)((unsigned)Fdim << 16),              // tensor_dim0[15:0] = 2048
               (int)(((unsigned)rows_avail & 0xFFFFu) << 16),  // tensor_dim1[15:0]
               (int)((unsigned)KC << 16),                // tile_dim0 = 128
               16,                                       // tile_dim1 = 16 rows (tile_dim2 = 0)
               Fdim,                                     // tensor_dim0_stride[31:0] = 2048
               0, 0 };
  i32x4 z4 = {0, 0, 0, 0};
#if __has_include(<hip/amd_detail/amd_gfx1250_TDM.h>)
  i32x8 z8 = {0, 0, 0, 0, 0, 0, 0, 0};
  __builtin_amdgcn_tensor_load_to_lds(g0, g1, z4, z4, z8, 0);   // 6-arg toolchain
#else
  __builtin_amdgcn_tensor_load_to_lds(g0, g1, z4, z4, 0);       // 5-arg toolchain (ROCm 7.2)
#endif
}
#endif

// ---------------- kernel 0: fp32 -> bf16 conversion ----------------
__global__ __launch_bounds__(256) void k_cvt(const float* __restrict__ src,
                                             __bf16* __restrict__ dst, int n) {
  int i = blockIdx.x * 256 + threadIdx.x;
  if (i < n) dst[i] = (__bf16)src[i];
}

// ---------------- kernel 1: h_linear = h @ W_hatt^T + b_hatt + b_att ----------------
__global__ __launch_bounds__(256) void k_hlin(const __bf16* __restrict__ bh,
                                              const __bf16* __restrict__ bWhatt,
                                              const float* __restrict__ b_hatt,
                                              const float* __restrict__ b_att,
                                              float* __restrict__ hlin) {
  int lane = threadIdx.x & 31;
  int w    = threadIdx.x >> 5;
  int m0   = blockIdx.x * 16;                // B rows
  int col0 = blockIdx.y * 128 + w * 16;      // H cols
  v8f acc = zero_v8f();
  for (int k0 = 0; k0 < Rdim; k0 += 32) {
    v16bf a  = load_a_bf(bh, Rdim, m0, k0, lane);
    v16bf bm = load_b_bf(bWhatt, Rdim, col0, k0, lane);
    acc = wmma_bf16(a, bm, acc);
  }
  int col = col0 + (lane & 15);
  float bias = b_hatt[col] + b_att[col];
#pragma unroll
  for (int r = 0; r < 8; ++r) {
    int row = m0 + r + ((lane >> 4) << 3);
    hlin[(size_t)row * Hdim + col] = acc[r] + bias;
  }
}

// ---------------- kernel 2: attention scores ----------------
// scores[b,n] = w_out . tanh(att_seq[b,n,:] @ W_att^T + hlin[b,:]) + b_out
__global__ __launch_bounds__(256) void k_scores(const float* __restrict__ att,
                                                const __bf16* __restrict__ bWatt,
                                                const float* __restrict__ hlin,
                                                const float* __restrict__ w_out,
                                                const float* __restrict__ b_out,
                                                float* __restrict__ scores) {
  int b    = blockIdx.x;
  int n0   = blockIdx.y * 16;
  int lane = threadIdx.x & 31;
  int w    = threadIdx.x >> 5;
  int col0 = w * 64;                          // each wave owns 64 of 512 H-cols

  v8f acc[4];
#pragma unroll
  for (int t = 0; t < 4; ++t) acc[t] = zero_v8f();

#if HAVE_TDM
  // Tensor Data Mover pipeline: wave 0 DMAs 16xKC fp32 A-chunks into LDS
  // (double buffered); all 8 waves consume via ds_load + cvt + wmma.
  __shared__ float abuf[2][16 * LDSROW];
  const float* gbase = att + (size_t)b * Nsq * Fdim + (size_t)n0 * Fdim;
  int rows_avail = Nsq - n0;                  // tail tile: OOB rows -> zeros
  unsigned lds0 = (unsigned)(size_t)&abuf[0][0];   // generic LDS ptr low 32 = LDS offset
  unsigned lds1 = (unsigned)(size_t)&abuf[1][0];
  bool issuer = (threadIdx.x < 32);
  if (issuer) tdm_issue(gbase, lds0, rows_avail);
  const int NCH = Fdim / KC;                  // 16 chunks
  for (int cidx = 0; cidx < NCH; ++cidx) {
    if (issuer) {
      if (cidx + 1 < NCH) {
        tdm_issue(gbase + (size_t)(cidx + 1) * KC, (cidx & 1) ? lds0 : lds1, rows_avail);
        __builtin_amdgcn_s_wait_tensorcnt(1);   // chunk cidx arrived, next in flight
      } else {
        __builtin_amdgcn_s_wait_tensorcnt(0);
      }
    }
    __syncthreads();                          // chunk cidx visible to all waves
    const float* buf = (cidx & 1) ? abuf[1] : abuf[0];
    int kg = cidx * KC;
#pragma unroll
    for (int kk = 0; kk < KC; kk += 32) {
      v16bf a = load_a_lds(buf, kk, lane);
#pragma unroll
      for (int t = 0; t < 4; ++t) {
        v16bf bm = load_b_bf(bWatt, Fdim, col0 + t * 16, kg + kk, lane);
        acc[t] = wmma_bf16(a, bm, acc[t]);
      }
    }
    __syncthreads();                          // before buffer is re-filled
  }
#else
  const float* Ab = att + (size_t)b * Nsq * Fdim;
  for (int k0 = 0; k0 < Fdim; k0 += 32) {
    v16bf a = load_a_f32(Ab, Fdim, n0, Nsq - 1, k0, lane);
#pragma unroll
    for (int t = 0; t < 4; ++t) {
      v16bf bm = load_b_bf(bWatt, Fdim, col0 + t * 16, k0, lane);
      acc[t] = wmma_bf16(a, bm, acc[t]);
    }
  }
#endif

  // epilogue: tanh + dot with w_out, reduce over H
  float partial[8];
#pragma unroll
  for (int r = 0; r < 8; ++r) partial[r] = 0.0f;
  const float* hl = hlin + (size_t)b * Hdim;
#pragma unroll
  for (int t = 0; t < 4; ++t) {
    int col = col0 + t * 16 + (lane & 15);
    float hb = hl[col];
    float wv = w_out[col];
#pragma unroll
    for (int r = 0; r < 8; ++r) {
      float s = tanhf(acc[t][r] + hb);
      partial[r] = fmaf(s, wv, partial[r]);
    }
  }
  // reduce across the 16 lanes of each half (masks stay within halves)
#pragma unroll
  for (int m = 1; m <= 8; m <<= 1) {
#pragma unroll
    for (int r = 0; r < 8; ++r) partial[r] += __shfl_xor(partial[r], m, 32);
  }
  __shared__ float part[8][16];
  if ((lane & 15) == 0) {
    int rb = (lane >> 4) * 8;                 // rows 0..7 (lane 0) / 8..15 (lane 16)
#pragma unroll
    for (int r = 0; r < 8; ++r) part[w][rb + r] = partial[r];
  }
  __syncthreads();
  if (threadIdx.x < 16) {                     // deterministic serial sum over 8 waves
    float s = b_out[0];
#pragma unroll
    for (int ww = 0; ww < 8; ++ww) s += part[ww][threadIdx.x];
    int n = n0 + threadIdx.x;
    if (n < Nsq) scores[b * NPAD + n] = s;
  }
}

// ---------------- kernel 3: softmax over N per batch ----------------
__global__ __launch_bounds__(256) void k_softmax(const float* __restrict__ scores,
                                                 float* __restrict__ cw) {
  int b = blockIdx.x, t = threadIdx.x;
  __shared__ float red[256];
  float v = (t < Nsq) ? scores[b * NPAD + t] : -3.4e38f;
  red[t] = v; __syncthreads();
#pragma unroll
  for (int s = 128; s > 0; s >>= 1) {
    if (t < s) red[t] = fmaxf(red[t], red[t + s]);
    __syncthreads();
  }
  float mx = red[0]; __syncthreads();
  float e = (t < Nsq) ? __expf(v - mx) : 0.0f;
  red[t] = e; __syncthreads();
#pragma unroll
  for (int s = 128; s > 0; s >>= 1) {
    if (t < s) red[t] += red[t + s];
    __syncthreads();
  }
  float inv = 1.0f / red[0];
  if (t < Nsq) cw[b * NPAD + t] = e * inv;
}

// ---------------- kernel 4: z[b,f] = sum_n att[b,n,f] * cw[b,n] (bf16 out) ----------------
__global__ __launch_bounds__(256) void k_z(const float* __restrict__ att,
                                           const float* __restrict__ cw,
                                           __bf16* __restrict__ zb) {
  int b  = blockIdx.x;
  int f0 = blockIdx.y * 512 + threadIdx.x * 2;
  __shared__ float scw[Nsq];
  if (threadIdx.x < Nsq) scw[threadIdx.x] = cw[b * NPAD + threadIdx.x];
  __syncthreads();
  const float* base = att + (size_t)b * Nsq * Fdim;
  float ax = 0.0f, ay = 0.0f;
  for (int n = 0; n < Nsq; ++n) {
    float wv = scw[n];
    float2 v = *(const float2*)(base + (size_t)n * Fdim + f0);
    ax = fmaf(wv, v.x, ax);
    ay = fmaf(wv, v.y, ay);
    int np = (n + 8 < Nsq) ? n + 8 : Nsq - 1;
    __builtin_prefetch(base + (size_t)np * Fdim + f0, 0, 0);  // global_prefetch_b8
  }
  zb[(size_t)b * Fdim + f0]     = (__bf16)ax;
  zb[(size_t)b * Fdim + f0 + 1] = (__bf16)ay;
}

// ---------------- kernel 5: all_input_sums = h@W_h2h^T + z@W_z2h^T + biases ----------------
__global__ __launch_bounds__(256) void k_gates_gemm(const __bf16* __restrict__ bh,
                                                    const __bf16* __restrict__ bz,
                                                    const __bf16* __restrict__ bWh2h,
                                                    const __bf16* __restrict__ bWz2h,
                                                    const float* __restrict__ b_h2h,
                                                    const float* __restrict__ b_z2h,
                                                    float* __restrict__ allsums) {
  int lane = threadIdx.x & 31;
  int w    = threadIdx.x >> 5;
  int col0 = blockIdx.x * 128 + w * 16;       // 4R = 4096 cols
  int m0   = blockIdx.y * 16;                 // B rows
  v8f acc = zero_v8f();
  for (int k0 = 0; k0 < Rdim; k0 += 32) {
    v16bf a  = load_a_bf(bh, Rdim, m0, k0, lane);
    v16bf bm = load_b_bf(bWh2h, Rdim, col0, k0, lane);
    acc = wmma_bf16(a, bm, acc);
  }
  for (int k0 = 0; k0 < Fdim; k0 += 32) {
    v16bf a  = load_a_bf(bz, Fdim, m0, k0, lane);
    v16bf bm = load_b_bf(bWz2h, Fdim, col0, k0, lane);
    acc = wmma_bf16(a, bm, acc);
  }
  int col = col0 + (lane & 15);
  float bias = b_h2h[col] + b_z2h[col];
#pragma unroll
  for (int r = 0; r < 8; ++r) {
    int row = m0 + r + ((lane >> 4) << 3);
    allsums[(size_t)row * (4 * Rdim) + col] = acc[r] + bias;
  }
}

// ---------------- kernel 6: LSTM gate elementwise ----------------
__global__ __launch_bounds__(256) void k_gate_ew(const float* __restrict__ allsums,
                                                 const float* __restrict__ cprev,
                                                 float* __restrict__ out) {
  int idx = blockIdx.x * 256 + threadIdx.x;   // 0 .. B*R-1
  int b = idx >> 10;                          // / R
  int j = idx & (Rdim - 1);
  const float* s = allsums + (size_t)b * (4 * Rdim);
  float ig = 1.0f / (1.0f + __expf(-s[j]));
  float fg = 1.0f / (1.0f + __expf(-s[Rdim + j]));
  float og = 1.0f / (1.0f + __expf(-s[2 * Rdim + j]));
  float gt = tanhf(s[3 * Rdim + j]);
  float nc = fg * cprev[idx] + ig * gt;
  float nh = og * tanhf(nc);
  out[idx] = nh;                              // next_h
  out[Bsz * Rdim + idx] = nc;                 // next_c
}

// ---------------- host launcher ----------------
extern "C" void kernel_launch(void* const* d_in, const int* in_sizes, int n_in,
                              void* d_out, int out_size, void* d_ws, size_t ws_size,
                              hipStream_t stream) {
  const float* att    = (const float*)d_in[0];
  const float* h      = (const float*)d_in[3];
  const float* cprev  = (const float*)d_in[4];
  const float* W_att  = (const float*)d_in[5];
  const float* b_att  = (const float*)d_in[6];
  const float* W_hatt = (const float*)d_in[7];
  const float* b_hatt = (const float*)d_in[8];
  const float* w_out  = (const float*)d_in[9];
  const float* b_out  = (const float*)d_in[10];
  const float* W_h2h  = (const float*)d_in[11];
  const float* b_h2h  = (const float*)d_in[12];
  const float* W_z2h  = (const float*)d_in[13];
  const float* b_z2h  = (const float*)d_in[14];
  float* out = (float*)d_out;

  // workspace layout (fp32 region then bf16 region, all 32B-aligned)
  float* ws_f    = (float*)d_ws;
  float* hlin    = ws_f;                       //  65536 f32
  float* scores  = ws_f + 65536;               //  26624 f32 (stride 208)
  float* cw      = ws_f + 92160;               //  26624 f32
  float* allsums = ws_f + 118784;              // 524288 f32
  __bf16* bfb    = (__bf16*)(ws_f + 643072);
  __bf16* bWatt  = bfb;                        // 512*2048
  __bf16* bWhatt = bWatt  + 1048576;           // 512*1024
  __bf16* bWh2h  = bWhatt + 524288;            // 4096*1024
  __bf16* bWz2h  = bWh2h  + 4194304;           // 4096*2048
  __bf16* bh     = bWz2h  + 8388608;           // 128*1024
  __bf16* bz     = bh     + 131072;            // 128*2048

  // 0) precision staging: weights + h -> bf16 (read once; L2-resident thereafter)
  k_cvt<<<(1048576 + 255) / 256, 256, 0, stream>>>(W_att,  bWatt,  1048576);
  k_cvt<<<(524288  + 255) / 256, 256, 0, stream>>>(W_hatt, bWhatt, 524288);
  k_cvt<<<(4194304 + 255) / 256, 256, 0, stream>>>(W_h2h,  bWh2h,  4194304);
  k_cvt<<<(8388608 + 255) / 256, 256, 0, stream>>>(W_z2h,  bWz2h,  8388608);
  k_cvt<<<(131072  + 255) / 256, 256, 0, stream>>>(h,      bh,     131072);

  // 1) h_linear (+ folded b_hatt + b_att)
  k_hlin<<<dim3(Bsz / 16, Hdim / 128), 256, 0, stream>>>(bh, bWhatt, b_hatt, b_att, hlin);

  // 2) attention scores (dominant GEMM, TDM-staged A stream)
  k_scores<<<dim3(Bsz, NPAD / 16), 256, 0, stream>>>(att, bWatt, hlin, w_out, b_out, scores);

  // 3) softmax over N
  k_softmax<<<Bsz, 256, 0, stream>>>(scores, cw);

  // 4) z = attention-weighted sum over N (bandwidth-bound; second att_seq pass)
  k_z<<<dim3(Bsz, Fdim / 512), 256, 0, stream>>>(att, cw, bz);

  // 5) gate pre-activations GEMM
  k_gates_gemm<<<dim3((4 * Rdim) / 128, Bsz / 16), 256, 0, stream>>>(
      bh, bz, bWh2h, bWz2h, b_h2h, b_z2h, allsums);

  // 6) LSTM elementwise -> d_out = [next_h | next_c]
  k_gate_ew<<<(Bsz * Rdim) / 256, 256, 0, stream>>>(allsums, cprev, out);
}
